// PointNeXt_3865470566875
// MI455X (gfx1250) — matrix operand order repaired and checked by reading, not compiled, not run on hardware
//
#include <hip/hip_runtime.h>
#include <hip/hip_bf16.h>

// ---------------------------------------------------------------------------
// PointNeXt forward for MI455X (gfx1250).
// GEMMs: v_wmma_f32_16x16x32_bf16, LDS-staged bf16 tiles, K-step 64
// (two WMMAs per barrier period). Staging loops use constant-stride 32-bit
// offsets + LDS store immediates so register pressure stays minimal.
// ---------------------------------------------------------------------------

#define BATCH 4
#define NPTS  8192
#define BN_EPS 1e-5f

typedef __attribute__((ext_vector_type(16))) __bf16 v16bf;
typedef __attribute__((ext_vector_type(8)))  __bf16 v8bf;
typedef __attribute__((ext_vector_type(8)))  float  v8f;

__device__ __forceinline__ __bf16 f2bf(float f) { return (__bf16)f; }

static inline int ceildiv(int a, int b) { return (a + b - 1) / b; }

// ---------------------------------------------------------------------------
// xyz extraction: x is (B,N,6) -> xyz (B,N,3)
// ---------------------------------------------------------------------------
__global__ void k_extract_xyz(const float* __restrict__ x, float* __restrict__ xyz, int n) {
  int t = blockIdx.x * blockDim.x + threadIdx.x;
  int total = BATCH * n;
  if (t >= total) return;
  xyz[t * 3 + 0] = x[t * 6 + 0];
  xyz[t * 3 + 1] = x[t * 6 + 1];
  xyz[t * 3 + 2] = x[t * 6 + 2];
}

// ---------------------------------------------------------------------------
// Farthest point sampling: one block per batch, iterative argmax.
// ---------------------------------------------------------------------------
__global__ void k_fps(const float* __restrict__ xyz, float* __restrict__ dist,
                      int* __restrict__ cent, int n, int npoint) {
  int b = blockIdx.x;
  int tid = threadIdx.x;
  const float* X = xyz + (size_t)b * n * 3;
  float* D = dist + (size_t)b * n;
  for (int i = tid; i < n; i += blockDim.x) D[i] = 1e10f;
  __syncthreads();

  __shared__ float smax[256];
  __shared__ int   simax[256];
  __shared__ int   sfar;
  int far = 0;
  for (int it = 0; it < npoint; ++it) {
    if (tid == 0) cent[b * npoint + it] = far;
    float cx = X[far * 3 + 0], cy = X[far * 3 + 1], cz = X[far * 3 + 2];
    float bmax = -1.0f; int bidx = 0;
    for (int i = tid; i < n; i += blockDim.x) {
      float dx = X[i * 3 + 0] - cx;
      float dy = X[i * 3 + 1] - cy;
      float dz = X[i * 3 + 2] - cz;
      float d = dx * dx + dy * dy + dz * dz;
      float nd = fminf(D[i], d);
      D[i] = nd;
      if (nd > bmax) { bmax = nd; bidx = i; }
    }
    smax[tid] = bmax; simax[tid] = bidx;
    __syncthreads();
    for (int s = 128; s > 0; s >>= 1) {
      if (tid < s) {
        if (smax[tid + s] > smax[tid]) { smax[tid] = smax[tid + s]; simax[tid] = simax[tid + s]; }
      }
      __syncthreads();
    }
    if (tid == 0) sfar = simax[0];
    __syncthreads();
    far = sfar;
    __syncthreads();
  }
}

// gather new_xyz[b,s,:] = xyz[b, cent[b,s], :]
__global__ void k_gather_xyz(const float* __restrict__ xyz, const int* __restrict__ cent,
                             float* __restrict__ nxyz, int n, int s) {
  int t = blockIdx.x * blockDim.x + threadIdx.x;
  int total = BATCH * s;
  if (t >= total) return;
  int b = t / s;
  int pid = cent[t];
  const float* p = xyz + ((size_t)b * n + pid) * 3;
  nxyz[t * 3 + 0] = p[0];
  nxyz[t * 3 + 1] = p[1];
  nxyz[t * 3 + 2] = p[2];
}

// ---------------------------------------------------------------------------
// kNN: one thread per query, insertion-sorted top-k (k <= 16), ascending dist.
// ---------------------------------------------------------------------------
__global__ void k_knn(const float* __restrict__ xyz, const float* __restrict__ q,
                      float* __restrict__ dout, int* __restrict__ iout,
                      int n, int s, int k) {
  int t = blockIdx.x * blockDim.x + threadIdx.x;
  int total = BATCH * s;
  if (t >= total) return;
  int b = t / s;
  const float* X = xyz + (size_t)b * n * 3;
  float qx = q[t * 3 + 0], qy = q[t * 3 + 1], qz = q[t * 3 + 2];
  float qn = qx * qx + qy * qy + qz * qz;
  float bd[16]; int bi[16];
  for (int j = 0; j < 16; ++j) { bd[j] = 3.0e38f; bi[j] = 0; }
  for (int i = 0; i < n; ++i) {
    float px = X[i * 3 + 0], py = X[i * 3 + 1], pz = X[i * 3 + 2];
    float d = qn + px * px + py * py + pz * pz - 2.0f * (qx * px + qy * py + qz * pz);
    if (d < bd[k - 1]) {
      int j = k - 1;
      while (j > 0 && d < bd[j - 1]) { bd[j] = bd[j - 1]; bi[j] = bi[j - 1]; --j; }
      bd[j] = d; bi[j] = i;
    }
  }
  for (int j = 0; j < k; ++j) { dout[(size_t)t * k + j] = bd[j]; iout[(size_t)t * k + j] = bi[j]; }
}

// grouping: out[b,s,k, 0:3]=xyz[idx]-new_xyz ; out[b,s,k, 3:3+c]=pts[idx]
__global__ void k_group(const float* __restrict__ xyz, const float* __restrict__ pts,
                        const float* __restrict__ nxyz, const int* __restrict__ idx,
                        float* __restrict__ out, int n, int s, int k, int c) {
  int t = blockIdx.x * blockDim.x + threadIdx.x;
  int total = BATCH * s * k;
  if (t >= total) return;
  int b = t / (s * k);
  int rem = t % (s * k);
  int si = rem / k;
  int pid = idx[t];
  const float* P3 = xyz + ((size_t)b * n + pid) * 3;
  const float* NX = nxyz + ((size_t)b * s + si) * 3;
  float* O = out + (size_t)t * (3 + c);
  O[0] = P3[0] - NX[0];
  O[1] = P3[1] - NX[1];
  O[2] = P3[2] - NX[2];
  const float* P = pts + ((size_t)b * n + pid) * c;
  for (int j = 0; j < c; ++j) O[3 + j] = P[j];
}

// ---------------------------------------------------------------------------
// WMMA GEMM: C[M,N] = A[M,K] * W[K,N] + bias (optional ReLU).
// Block = 128 threads (4 waves). Block tile 64x16; each wave one 16x16 tile.
// K-step 64: 64x64 A tile (8KB) + 16x64 B tile (2KB) staged in LDS as bf16,
// two v_wmma_f32_16x16x32_bf16 per barrier period.
// Staging index algebra keeps live state tiny:
//   A: thread k-offset (tid&63) constant; row advances by 2/iter ->
//      global offset base + i*2K (32-bit), LDS offset base + i*256B (imm).
//   B: thread column (tid&15) constant; k advances by 8/iter ->
//      global offset base + i*8N, LDS offset base + i*16B (imm).
// Fragment reads are contiguous ds_load_b128s per the CDNA5 VGPR layouts.
// ---------------------------------------------------------------------------
__global__ void k_gemm_bias(const float* __restrict__ A, const float* __restrict__ W,
                            const float* __restrict__ bias, float* __restrict__ C,
                            int M, int K, int N, int relu) {
  __shared__ __align__(16) __bf16 As[64 * 64]; // [row][k], 8 KB
  __shared__ __align__(16) __bf16 Bs[16 * 64]; // [col][k] (K-transposed), 2 KB

  int tid  = threadIdx.x;      // 0..127
  int wave = tid >> 5;
  int lane = tid & 31;
  int half = lane >> 4;
  int mr   = lane & 15;
  int rowbase = blockIdx.x * 64;
  int col0    = blockIdx.y * 16;
  int col     = col0 + mr;

  // per-thread constant staging coordinates
  int a_cc = tid & 63;         // k offset within A tile (constant)
  int a_r0 = tid >> 6;         // 0..1, row advances by 2 per iteration
  int a_lds0 = a_r0 * 64 + a_cc;
  int b_nn = tid & 15;         // column within B tile (constant)
  int b_k0 = tid >> 4;         // 0..7, k advances by 8 per iteration
  int b_lds0 = b_nn * 64 + b_k0;
  int b_gn = col0 + b_nn;
  bool b_nok = (b_gn < N);

  v8f acc = {0.f, 0.f, 0.f, 0.f, 0.f, 0.f, 0.f, 0.f};

  for (int kb = 0; kb < K; kb += 64) {
    // --- A tile: 64x64, branch-free clamped loads, constant-stride offsets
    {
      int gk = kb + a_cc;
      bool kok = (gk < K);
      unsigned off0 = (unsigned)(rowbase + a_r0) * (unsigned)K + (unsigned)gk;
#pragma unroll 4
      for (int i = 0; i < 32; ++i) {
        int gr = rowbase + a_r0 + 2 * i;
        bool inb = kok && (gr < M);
        unsigned aoff = off0 + (unsigned)(2 * i) * (unsigned)K;
        float v = A[inb ? aoff : 0u];
        As[a_lds0 + i * 128] = f2bf(inb ? v : 0.0f);
      }
    }
    // --- B tile: 64(k) x 16(n), stored transposed Bs[n][k]
    {
#pragma unroll 4
      for (int i = 0; i < 8; ++i) {
        int gk = kb + b_k0 + 8 * i;
        bool inb = b_nok && (gk < K);
        unsigned woff = (unsigned)gk * (unsigned)N + (unsigned)b_gn;
        float v = W[inb ? woff : 0u];
        Bs[b_lds0 + i * 8] = f2bf(inb ? v : 0.0f);
      }
    }
    __syncthreads();

    const __bf16* Ar = &As[(wave * 16 + mr) * 64];
    const __bf16* Bc = &Bs[mr * 64];
#pragma unroll
    for (int s = 0; s < 2; ++s) {
      union { v16bf v; v8bf h[2]; } av, bv;
      av.h[0] = *(const v8bf*)(Ar + 32 * s + 8 * half);
      av.h[1] = *(const v8bf*)(Ar + 32 * s + 16 + 8 * half);
      bv.h[0] = *(const v8bf*)(Bc + 32 * s + 8 * half);
      bv.h[1] = *(const v8bf*)(Bc + 32 * s + 16 + 8 * half);
      acc = __builtin_amdgcn_wmma_f32_16x16x32_bf16(
          /*neg_a=*/false, av.v, /*neg_b=*/false, bv.v,
          /*c_mod=*/(short)0, acc, /*reuse_a=*/false, /*reuse_b=*/false);
    }
    __syncthreads();
  }

  float bval = (col < N) ? bias[col] : 0.0f;
#pragma unroll
  for (int r = 0; r < 8; ++r) {
    int m = rowbase + wave * 16 + half * 8 + r;
    if (m < M && col < N) {
      float v = acc[r] + bval;
      if (relu) v = fmaxf(v, 0.0f);
      C[(size_t)m * N + col] = v;
    }
  }
}

// per-column mean/var over M rows (one block per column)
__global__ void k_colstats(const float* __restrict__ X, float* __restrict__ mean,
                           float* __restrict__ var, int M, int N) {
  int col = blockIdx.x;
  __shared__ float ss[256], sq[256];
  float s = 0.f, q = 0.f;
  for (int i = threadIdx.x; i < M; i += blockDim.x) {
    float v = X[(size_t)i * N + col];
    s += v; q += v * v;
  }
  ss[threadIdx.x] = s; sq[threadIdx.x] = q;
  __syncthreads();
  for (int st = 128; st > 0; st >>= 1) {
    if (threadIdx.x < st) { ss[threadIdx.x] += ss[threadIdx.x + st]; sq[threadIdx.x] += sq[threadIdx.x + st]; }
    __syncthreads();
  }
  if (threadIdx.x == 0) {
    float mu = ss[0] / (float)M;
    mean[col] = mu;
    var[col] = sq[0] / (float)M - mu * mu;
  }
}

// in-place BN apply + optional residual add + optional ReLU
__global__ void k_bn(float* __restrict__ X, const float* __restrict__ mean,
                     const float* __restrict__ var, const float* __restrict__ g,
                     const float* __restrict__ beta, const float* __restrict__ addsrc,
                     int M, int N, int relu) {
  int t = blockIdx.x * blockDim.x + threadIdx.x;
  if (t >= M * N) return;
  int c = t % N;
  float v = (X[t] - mean[c]) * rsqrtf(var[c] + BN_EPS) * g[c] + beta[c];
  if (addsrc) v += addsrc[t];
  if (relu) v = fmaxf(v, 0.0f);
  X[t] = v;
}

// max over the k neighbor axis: (b,s,k,c) -> (b,s,c)
__global__ void k_maxpool(const float* __restrict__ F, float* __restrict__ out,
                          int s, int k, int c) {
  int t = blockIdx.x * blockDim.x + threadIdx.x;
  int total = BATCH * s * c;
  if (t >= total) return;
  int bs = t / c;
  int ci = t % c;
  const float* base = F + ((size_t)bs * k) * c + ci;
  float m = base[0];
  for (int j = 1; j < k; ++j) m = fmaxf(m, base[(size_t)j * c]);
  out[t] = m;
}

// 3-NN inverse-distance interpolation of pts1 -> columns [0,c1) of out (ld-stride rows)
__global__ void k_interp(const float* __restrict__ pts1, const float* __restrict__ dist,
                         const int* __restrict__ idx, float* __restrict__ out,
                         int n1, int s, int c1, int ld) {
  int t = blockIdx.x * blockDim.x + threadIdx.x;
  int total = BATCH * s * c1;
  if (t >= total) return;
  int bs = t / c1;
  int ci = t % c1;
  int b = bs / s;
  float d0 = fmaxf(dist[bs * 3 + 0], 1e-10f);
  float d1 = fmaxf(dist[bs * 3 + 1], 1e-10f);
  float d2 = fmaxf(dist[bs * 3 + 2], 1e-10f);
  float w0 = 1.0f / (d0 + 1e-8f);
  float w1 = 1.0f / (d1 + 1e-8f);
  float w2 = 1.0f / (d2 + 1e-8f);
  float ws = w0 + w1 + w2;
  w0 /= ws; w1 /= ws; w2 /= ws;
  const float* P = pts1 + (size_t)b * n1 * c1;
  float v = w0 * P[(size_t)idx[bs * 3 + 0] * c1 + ci]
          + w1 * P[(size_t)idx[bs * 3 + 1] * c1 + ci]
          + w2 * P[(size_t)idx[bs * 3 + 2] * c1 + ci];
  out[(size_t)bs * ld + ci] = v;
}

// copy (rows,c) src into dst columns [off, off+c) with row stride ld
__global__ void k_copycols(const float* __restrict__ src, float* __restrict__ dst,
                           int rows, int c, int ld, int off) {
  int t = blockIdx.x * blockDim.x + threadIdx.x;
  if (t >= rows * c) return;
  int r = t / c;
  int ci = t % c;
  dst[(size_t)r * ld + off + ci] = src[(size_t)r * c + ci];
}

// ---------------------------------------------------------------------------
// Host-side helpers
// ---------------------------------------------------------------------------
static void gemm(const float* A, const float* W, const float* bias, float* C,
                 int M, int K, int N, int relu, hipStream_t st) {
  dim3 g(ceildiv(M, 64), ceildiv(N, 16));
  k_gemm_bias<<<g, 128, 0, st>>>(A, W, bias, C, M, K, N, relu);
}

static void bn_stats_apply(float* X, int M, int N, const float* beta, const float* g,
                           float* meanb, float* varb, const float* add, int relu,
                           hipStream_t st) {
  k_colstats<<<N, 256, 0, st>>>(X, meanb, varb, M, N);
  int tot = M * N;
  k_bn<<<ceildiv(tot, 256), 256, 0, st>>>(X, meanb, varb, g, beta, add, M, N, relu);
}

struct Scratch {
  float* fpsdist; int* cent;
  float* knnd; int* knni;
  float* G; float* SC; float* H1; float* H2; float* H3;
  float* meanb; float* varb;
};

// Parameter flattening (JAX pytree: dict keys sorted alphabetically per level):
//  0: cls1.W   1: cls1.b   2: cls2.W   3: cls2.b
//  fp block (base..base+7): bn1.beta, bn1.g, bn2.beta, bn2.g, fc1.W, fc1.b, fc2.W, fc2.b
//  sa block (base..base+15): bn1.beta, bn1.g, bn2.beta, bn2.g, bn3.beta, bn3.g,
//    fc1.W, fc1.b, fc2.W, fc2.b, fc3.W, fc3.b, sc_bn.beta, sc_bn.g, sc_fc.W, sc_fc.b
//  76: x
#define PF(i) ((const float*)d_in[(i)])

static void set_abstraction(void* const* d_in, int pbase,
                            const float* xyz, const float* pts, int n, int c,
                            int npoint, int k, int cout,
                            float* new_xyz, float* pooled,
                            const Scratch& S, hipStream_t st) {
  int cin = c + 3;
  int ec = 2 * cin;
  k_fps<<<BATCH, 256, 0, st>>>(xyz, S.fpsdist, S.cent, n, npoint);
  k_gather_xyz<<<ceildiv(BATCH * npoint, 256), 256, 0, st>>>(xyz, S.cent, new_xyz, n, npoint);
  k_knn<<<ceildiv(BATCH * npoint, 128), 128, 0, st>>>(xyz, new_xyz, S.knnd, S.knni, n, npoint, k);
  int Mrows = BATCH * npoint * k;
  k_group<<<ceildiv(Mrows, 256), 256, 0, st>>>(xyz, pts, new_xyz, S.knni, S.G, n, npoint, k, c);

  const float* bn1b = PF(pbase + 0);  const float* bn1g = PF(pbase + 1);
  const float* bn2b = PF(pbase + 2);  const float* bn2g = PF(pbase + 3);
  const float* bn3b = PF(pbase + 4);  const float* bn3g = PF(pbase + 5);
  const float* fc1W = PF(pbase + 6);  const float* fc1b = PF(pbase + 7);
  const float* fc2W = PF(pbase + 8);  const float* fc2b = PF(pbase + 9);
  const float* fc3W = PF(pbase + 10); const float* fc3b = PF(pbase + 11);
  const float* scbb = PF(pbase + 12); const float* scbg = PF(pbase + 13);
  const float* scfW = PF(pbase + 14); const float* scfb = PF(pbase + 15);

  // shortcut branch: sc = BN(linear(G))
  gemm(S.G, scfW, scfb, S.SC, Mrows, cin, cout, 0, st);
  bn_stats_apply(S.SC, Mrows, cout, scbb, scbg, S.meanb, S.varb, nullptr, 0, st);
  // main branch
  gemm(S.G, fc1W, fc1b, S.H1, Mrows, cin, ec, 0, st);
  bn_stats_apply(S.H1, Mrows, ec, bn1b, bn1g, S.meanb, S.varb, nullptr, 1, st);
  gemm(S.H1, fc2W, fc2b, S.H2, Mrows, ec, ec, 0, st);
  bn_stats_apply(S.H2, Mrows, ec, bn2b, bn2g, S.meanb, S.varb, nullptr, 1, st);
  gemm(S.H2, fc3W, fc3b, S.H3, Mrows, ec, cout, 0, st);
  // relu(BN(h) + sc)
  bn_stats_apply(S.H3, Mrows, cout, bn3b, bn3g, S.meanb, S.varb, S.SC, 1, st);
  k_maxpool<<<ceildiv(BATCH * npoint * cout, 256), 256, 0, st>>>(S.H3, pooled, npoint, k, cout);
}

static void feature_prop(void* const* d_in, int pbase,
                         const float* xyz1 /*coarse n1*/, const float* xyz2 /*fine n2*/,
                         const float* pts1, int c1, const float* pts2, int c2,
                         float* outpts, int cout, int n1, int n2,
                         const Scratch& S, hipStream_t st) {
  k_knn<<<ceildiv(BATCH * n2, 128), 128, 0, st>>>(xyz1, xyz2, S.knnd, S.knni, n1, n2, 3);
  int rows = BATCH * n2;
  int ld = c1 + c2;
  k_interp<<<ceildiv(rows * c1, 256), 256, 0, st>>>(pts1, S.knnd, S.knni, S.G, n1, n2, c1, ld);
  k_copycols<<<ceildiv(rows * c2, 256), 256, 0, st>>>(pts2, S.G, rows, c2, ld, c1);

  const float* bn1b = PF(pbase + 0); const float* bn1g = PF(pbase + 1);
  const float* bn2b = PF(pbase + 2); const float* bn2g = PF(pbase + 3);
  const float* fc1W = PF(pbase + 4); const float* fc1b = PF(pbase + 5);
  const float* fc2W = PF(pbase + 6); const float* fc2b = PF(pbase + 7);

  gemm(S.G, fc1W, fc1b, S.H1, rows, ld, cout, 0, st);
  bn_stats_apply(S.H1, rows, cout, bn1b, bn1g, S.meanb, S.varb, nullptr, 1, st);
  gemm(S.H1, fc2W, fc2b, outpts, rows, cout, cout, 0, st);
  bn_stats_apply(outpts, rows, cout, bn2b, bn2g, S.meanb, S.varb, nullptr, 1, st);
}

extern "C" void kernel_launch(void* const* d_in, const int* in_sizes, int n_in,
                              void* d_out, int out_size, void* d_ws, size_t ws_size,
                              hipStream_t stream) {
  (void)in_sizes; (void)n_in; (void)out_size; (void)ws_size;
  const float* x = PF(76); // (B, N, 6)

  // bump allocator over d_ws
  char* base = (char*)d_ws;
  size_t off = 0;
  auto allocf = [&](size_t nf) -> float* { float* p = (float*)(base + off); off += nf * sizeof(float); return p; };
  auto alloci = [&](size_t ni) -> int*   { int*   p = (int*)(base + off);   off += ni * sizeof(int);   return p; };

  // persistent buffers
  float* xyz  = allocf((size_t)BATCH * NPTS * 3);
  float* xyz1 = allocf((size_t)BATCH * 2048 * 3);
  float* xyz2 = allocf((size_t)BATCH * 512 * 3);
  float* xyz3 = allocf((size_t)BATCH * 128 * 3);
  float* p1   = allocf((size_t)BATCH * 2048 * 128);
  float* p2   = allocf((size_t)BATCH * 512 * 256);
  float* p3   = allocf((size_t)BATCH * 128 * 512);
  float* l2   = allocf((size_t)BATCH * 512 * 256);
  float* l1   = allocf((size_t)BATCH * 2048 * 128);
  float* l0   = allocf((size_t)BATCH * NPTS * 64);

  // reusable scratch (sized for the max across stages)
  Scratch S;
  S.fpsdist = allocf((size_t)BATCH * NPTS);
  S.cent    = alloci((size_t)BATCH * 2048);
  S.knnd    = allocf((size_t)BATCH * 2048 * 16);
  S.knni    = alloci((size_t)BATCH * 2048 * 16);
  S.G       = allocf((size_t)4500000);   // max grouped/concat rows*cols (fp1: 32768*134)
  S.SC      = allocf((size_t)16777216);  // sa1: 131072*128
  S.H1      = allocf((size_t)8585216);   // sa2: 32768*262
  S.H2      = allocf((size_t)8585216);
  S.H3      = allocf((size_t)16777216);
  S.meanb   = allocf(1024);
  S.varb    = allocf(1024);

  k_extract_xyz<<<ceildiv(BATCH * NPTS, 256), 256, 0, stream>>>(x, xyz, NPTS);

  // set-abstraction stages (param bases per flattening note above)
  set_abstraction(d_in, 28, xyz,  x,  NPTS, 6,   2048, 16, 128, xyz1, p1, S, stream); // sa1
  set_abstraction(d_in, 44, xyz1, p1, 2048, 128, 512,  16, 256, xyz2, p2, S, stream); // sa2
  set_abstraction(d_in, 60, xyz2, p2, 512,  256, 128,  16, 512, xyz3, p3, S, stream); // sa3

  // feature propagation
  feature_prop(d_in, 20, xyz3, xyz2, p3, 512, p2, 256, l2, 256, 128,  512,  S, stream); // fp3
  feature_prop(d_in, 12, xyz2, xyz1, l2, 256, p1, 128, l1, 128, 512,  2048, S, stream); // fp2
  feature_prop(d_in, 4,  xyz1, xyz,  l1, 128, x,  6,   l0, 64,  2048, NPTS, S, stream); // fp1

  // classifier head: relu(l0 @ W1 + b1) @ W2 + b2 -> (B,N,2)
  float* clsh = S.H1; // free at this point
  gemm(l0, PF(0), PF(1), clsh, BATCH * NPTS, 64, 64, 1, stream);
  gemm(clsh, PF(2), PF(3), (float*)d_out, BATCH * NPTS, 64, 2, 0, stream);
}
